// GroupedMultiQueryAttention_60756607369638
// MI455X (gfx1250) — compile-verified, matrix operand order checked by reading, not run on hardware
//
#include <hip/hip_runtime.h>
#include <hip/hip_bf16.h>

typedef __bf16 bf16;
typedef __attribute__((ext_vector_type(16))) __bf16 v16bf;
typedef __attribute__((ext_vector_type(8)))  float  v8f;
typedef __attribute__((ext_vector_type(4)))  unsigned int u32x4;
typedef __attribute__((ext_vector_type(4)))  int          i32x4;
typedef __attribute__((ext_vector_type(8)))  int          i32x8;

#define WMMA_BF16(a, b, c) \
  __builtin_amdgcn_wmma_f32_16x16x32_bf16(false, (a), false, (b), (short)0, (c), false, false)

// ---------------------------------------------------------------------------
// Fragment loaders (CDNA5 wave32 WMMA layouts, CDNA5 ISA §7.12.2)
// A fragment, 16x32 bf16: lane L(0..15) -> row L, K = {0..7,16..23}+8*half
// ---------------------------------------------------------------------------
__device__ __forceinline__ v16bf load_frag_k32(const bf16* p, int ld, int row0,
                                               int k0, int lane) {
  const int half = (lane >> 4) & 1;
  const bf16* base = p + (size_t)(row0 + (lane & 15)) * ld + k0 + half * 8;
  v16bf a;
#pragma unroll
  for (int i = 0; i < 8; ++i) {
    a[i]     = base[i];        // K = k0 + 8*half + i
    a[8 + i] = base[16 + i];   // K = k0 + 16 + 8*half + i
  }
  return a;
}

// B fragment for B[k][n] stored row-major in k (leading dim = ld), column-strided.
__device__ __forceinline__ v16bf load_fragB_strided(const bf16* p, int ld, int k0,
                                                    int n0, int lane) {
  const int half = (lane >> 4) & 1;
  const bf16* base = p + (size_t)(k0 + half * 8) * ld + n0 + (lane & 15);
  v16bf a;
#pragma unroll
  for (int i = 0; i < 8; ++i) {
    a[i]     = base[(size_t)i * ld];
    a[8 + i] = base[(size_t)(16 + i) * ld];
  }
  return a;
}

// ---------------------------------------------------------------------------
// Tensor Data Mover: DMA a 2D tile (tile_x elems wide, tile_y rows, 2-byte
// elements, row stride `row_stride` elems) from global -> LDS.
// D# bitfields per CDNA5 ISA. This toolchain
// exposes the 6-arg builtin: (g0, g1, g2, g3, g4, cpol); groups 2..4 zero
// for a plain 2D tensor.
// ---------------------------------------------------------------------------
__device__ __forceinline__ void tdm_load_2d(unsigned lds_off,
                                            unsigned long long gaddr,
                                            unsigned tile_x, unsigned tile_y,
                                            unsigned row_stride) {
  u32x4 g0;
  g0[0] = 1u;                                               // count=1 (valid user D#)
  g0[1] = lds_off;                                          // lds_addr [63:32]
  g0[2] = (unsigned)(gaddr & 0xFFFFFFFFull);                // global_addr [95:64]
  g0[3] = (unsigned)((gaddr >> 32) & 0x1FFFFFFull)          // global_addr [120:96]
          | (2u << 30);                                     // type=2 ("image")
  i32x8 g1;
  g1[0] = (int)(1u << 16);                                  // wg_mask=0, data_size=1 (2B)
  g1[1] = (int)((tile_x & 0xFFFFu) << 16);                  // tensor_dim0[15:0]
  g1[2] = (int)(((tile_x >> 16) & 0xFFFFu)                  // tensor_dim0[31:16]
          | (0xFFFFu << 16));                               // tensor_dim1[15:0] (huge)
  g1[3] = (int)(0x7FFFu                                     // tensor_dim1[31:16]
          | ((tile_x & 0xFFFFu) << 16));                    // tile_dim0
  g1[4] = (int)(tile_y & 0xFFFFu);                          // tile_dim1 (tile_dim2=0)
  g1[5] = (int)row_stride;                                  // tensor_dim0_stride[31:0]
  g1[6] = 0;                                                // stride hi, dim1_stride lo
  g1[7] = 0;
  i32x4 z4 = {0, 0, 0, 0};
  i32x8 z8 = {0, 0, 0, 0, 0, 0, 0, 0};
  __builtin_amdgcn_tensor_load_to_lds(g0, g1, z4, z4, z8, 0);
}

__device__ __forceinline__ unsigned lds_offset_of(const void* p) {
  // Generic pointer to LDS: ADDR[31:0] is the LDS byte address (ISA §10.2).
  return (unsigned)(unsigned long long)p;
}

// ---------------------------------------------------------------------------
// Problem constants: b=4, s=2048, d_model=2048, n_q=32, n_kv=8, g=4, d_head=64
// ---------------------------------------------------------------------------
#define S 2048
#define B 4
#define DM 2048
#define NKV 8
#define DH 64
#define ROWS (B * S)          // 8192
#define NQKV (3 * NKV * DH)   // 1536
#define NZ (NKV * DH)         // 512

// ---- prep: fp32 -> bf16 cast of activations ----
__global__ void cast_x_bf16(const float* __restrict__ in, bf16* __restrict__ out,
                            size_t n) {
  size_t i = (size_t)blockIdx.x * blockDim.x + threadIdx.x;
  if (i < n) out[i] = (bf16)in[i];
}

// ---- prep: transposed bf16 QKV weight [1536 x 2048], Q group-summed ----
__global__ void build_bqkv(const float* __restrict__ WQ, const float* __restrict__ WK,
                           const float* __restrict__ WV, bf16* __restrict__ Bt) {
  int idx = blockIdx.x * blockDim.x + threadIdx.x;
  if (idx >= NQKV * DM) return;
  int j = idx / DM, k = idx % DM;
  float v;
  if (j < NKV * DH) {                       // summed Q weights
    int n = j / DH, h = j % DH;
    float sacc = 0.f;
#pragma unroll
    for (int gg = 0; gg < 4; ++gg)
      sacc += WQ[(size_t)(n * 4 + gg) * DM * DH + (size_t)k * DH + h];
    v = sacc;
  } else if (j < 2 * NKV * DH) {
    int jj = j - NKV * DH;
    v = WK[(size_t)(jj / DH) * DM * DH + (size_t)k * DH + (jj % DH)];
  } else {
    int jj = j - 2 * NKV * DH;
    v = WV[(size_t)(jj / DH) * DM * DH + (size_t)k * DH + (jj % DH)];
  }
  Bt[(size_t)j * DM + k] = (bf16)v;
}

// ---- prep: transposed bf16 output weight [2048 x 512] ----
__global__ void build_bo(const float* __restrict__ WO, bf16* __restrict__ Bt) {
  int idx = blockIdx.x * blockDim.x + threadIdx.x;
  if (idx >= DM * NZ) return;
  int d = idx / NZ, j = idx % NZ;   // j = n*64+h ; WO[n][h][d]
  Bt[(size_t)d * NZ + j] = (bf16)WO[(size_t)j * DM + d];
}

// ---------------------------------------------------------------------------
// GEMM 1: qkv[8192 x 1536](bf16) = x_bf[8192 x 2048] @ Bqkv^T
// ---------------------------------------------------------------------------
__global__ void __launch_bounds__(128) gemm_qkv(const bf16* __restrict__ A,
                                                const bf16* __restrict__ Bt,
                                                bf16* __restrict__ C) {
  const int lane = threadIdx.x & 31;
  const int wid = threadIdx.x >> 5;
  const int m0 = blockIdx.x * 16;
  const int n0 = (blockIdx.y * 4 + wid) * 64;
  v8f c0 = {}, c1 = {}, c2 = {}, c3 = {};
  for (int k0 = 0; k0 < DM; k0 += 32) {
    v16bf a  = load_frag_k32(A, DM, m0, k0, lane);
    v16bf b0 = load_frag_k32(Bt, DM, n0,      k0, lane);
    v16bf b1 = load_frag_k32(Bt, DM, n0 + 16, k0, lane);
    v16bf b2 = load_frag_k32(Bt, DM, n0 + 32, k0, lane);
    v16bf b3 = load_frag_k32(Bt, DM, n0 + 48, k0, lane);
    c0 = WMMA_BF16(a, b0, c0);
    c1 = WMMA_BF16(a, b1, c1);
    c2 = WMMA_BF16(a, b2, c2);
    c3 = WMMA_BF16(a, b3, c3);
  }
  const int half = (lane >> 4) & 1, lrow = lane & 15;
#pragma unroll
  for (int r = 0; r < 8; ++r) {
    size_t row = (size_t)(m0 + half * 8 + r);
    C[row * NQKV + n0 +      lrow] = (bf16)c0[r];
    C[row * NQKV + n0 + 16 + lrow] = (bf16)c1[r];
    C[row * NQKV + n0 + 32 + lrow] = (bf16)c2[r];
    C[row * NQKV + n0 + 48 + lrow] = (bf16)c3[r];
  }
}

// ---------------------------------------------------------------------------
// Flash attention, block = 4 waves = 64 query rows of one (batch, head).
// Per 32-key chunk: wave 0 DMAs K-chunk + V-chunk (32x64 bf16 each) into LDS
// via the Tensor Data Mover, waits TENSORcnt, block barrier; all waves then
// run S = Q K^T (WMMA from LDS), online softmax, P staged through LDS,
// O += P V (WMMA, V fragments as strided ds reads). kv trip count is
// block-uniform (causal bound of the last wave); fully-masked chunks are
// inert because m is finite after chunk 0 (key 0 is never masked).
// ---------------------------------------------------------------------------
__global__ void __launch_bounds__(128) flash_attn(const bf16* __restrict__ qkv,
                                                  bf16* __restrict__ Z) {
  const int lane = threadIdx.x & 31;
  const int wid = threadIdx.x >> 5;
  const int qtile = blockIdx.x * 4 + wid;   // 0..127
  const int head = blockIdx.y;              // 0..7
  const int batch = blockIdx.z;             // 0..3
  const int ld = NQKV;
  const bf16* Qp = qkv + (size_t)batch * S * ld + head * DH;
  const bf16* Kp = Qp + NKV * DH;
  const bf16* Vp = Qp + 2 * NKV * DH;

  __shared__ bf16 ldsK[32 * DH];            // 4 KB, shared by block
  __shared__ bf16 ldsV[32 * DH];            // 4 KB, shared by block
  __shared__ bf16 lds_p[4][16 * 32];        // 1 KB per wave
  bf16* Pst = lds_p[wid];

  const int half = (lane >> 4) & 1, lrow = lane & 15;
  const int q0 = qtile * 16;

  v16bf qa0 = load_frag_k32(Qp, ld, q0, 0, lane);
  v16bf qa1 = load_frag_k32(Qp, ld, q0, 32, lane);

  v8f o0 = {}, o1 = {}, o2 = {}, o3 = {};
  float m[8], l[8];
#pragma unroll
  for (int r = 0; r < 8; ++r) { m[r] = -1e30f; l[r] = 0.f; }

  const unsigned kOff = lds_offset_of((const void*)ldsK);
  const unsigned vOff = lds_offset_of((const void*)ldsV);
  const int qend_blk = (blockIdx.x + 1) * 64;   // block-uniform causal bound

  for (int kv0 = 0; kv0 < qend_blk; kv0 += 32) {
    __syncthreads();                        // prior chunk fully consumed
    if (wid == 0) {
      tdm_load_2d(kOff, (unsigned long long)(const void*)(Kp + (size_t)kv0 * ld),
                  DH, 32, ld);
      tdm_load_2d(vOff, (unsigned long long)(const void*)(Vp + (size_t)kv0 * ld),
                  DH, 32, ld);
      __builtin_amdgcn_s_wait_tensorcnt(0);
    }
    __syncthreads();                        // staged K/V visible to all waves

    // ---- scores: two 16x16 tiles over keys [kv0, kv0+32), frags from LDS ----
    v16bf kb00 = load_frag_k32(ldsK, DH, 0,  0,  lane);
    v16bf kb01 = load_frag_k32(ldsK, DH, 0,  32, lane);
    v16bf kb10 = load_frag_k32(ldsK, DH, 16, 0,  lane);
    v16bf kb11 = load_frag_k32(ldsK, DH, 16, 32, lane);
    v8f s0 = {}, s1 = {};
    s0 = WMMA_BF16(qa0, kb00, s0);
    s0 = WMMA_BF16(qa1, kb01, s0);
    s1 = WMMA_BF16(qa0, kb10, s1);
    s1 = WMMA_BF16(qa1, kb11, s1);

    // ---- scale + causal mask + online softmax ----
#pragma unroll
    for (int r = 0; r < 8; ++r) {
      const int qrow = q0 + half * 8 + r;
      float e0 = s0[r] * 0.125f;            // 1/sqrt(64)
      float e1 = s1[r] * 0.125f;
      if (kv0 + lrow > qrow) e0 = -1e30f;
      if (kv0 + 16 + lrow > qrow) e1 = -1e30f;

      float mx = fmaxf(e0, e1);
#pragma unroll
      for (int off = 8; off >= 1; off >>= 1) mx = fmaxf(mx, __shfl_xor(mx, off, 32));
      const float mnew = fmaxf(m[r], mx);
      const float alpha = __expf(m[r] - mnew);
      const float p0 = __expf(e0 - mnew);
      const float p1 = __expf(e1 - mnew);
      float rs = p0 + p1;
#pragma unroll
      for (int off = 8; off >= 1; off >>= 1) rs += __shfl_xor(rs, off, 32);
      l[r] = l[r] * alpha + rs;
      m[r] = mnew;
      o0[r] *= alpha; o1[r] *= alpha; o2[r] *= alpha; o3[r] *= alpha;

      const int prow = half * 8 + r;        // C-layout -> LDS row-major
      Pst[prow * 32 + lrow]      = (bf16)p0;
      Pst[prow * 32 + 16 + lrow] = (bf16)p1;
    }
    asm volatile("s_wait_dscnt 0" ::: "memory");   // DS store -> DS load (same wave)

    // ---- O += P(16x32) @ V(32x64), all operands from LDS ----
    v16bf pa  = load_frag_k32(Pst, 32, 0, 0, lane);
    v16bf vb0 = load_fragB_strided(ldsV, DH, 0, 0,  lane);
    v16bf vb1 = load_fragB_strided(ldsV, DH, 0, 16, lane);
    v16bf vb2 = load_fragB_strided(ldsV, DH, 0, 32, lane);
    v16bf vb3 = load_fragB_strided(ldsV, DH, 0, 48, lane);
    o0 = WMMA_BF16(pa, vb0, o0);
    o1 = WMMA_BF16(pa, vb1, o1);
    o2 = WMMA_BF16(pa, vb2, o2);
    o3 = WMMA_BF16(pa, vb3, o3);
  }

  // ---- epilogue: normalize and store Z (bf16) ----
#pragma unroll
  for (int r = 0; r < 8; ++r) {
    const float inv = 1.0f / l[r];
    const size_t row = (size_t)batch * S + q0 + half * 8 + r;
    const size_t zoff = row * NZ + head * DH;
    Z[zoff +      lrow] = (bf16)(o0[r] * inv);
    Z[zoff + 16 + lrow] = (bf16)(o1[r] * inv);
    Z[zoff + 32 + lrow] = (bf16)(o2[r] * inv);
    Z[zoff + 48 + lrow] = (bf16)(o3[r] * inv);
  }
}

// ---------------------------------------------------------------------------
// GEMM 2: out[8192 x 2048](f32) = Z[8192 x 512] @ W_O ; Bt_o is [2048 x 512]
// ---------------------------------------------------------------------------
__global__ void __launch_bounds__(128) gemm_out(const bf16* __restrict__ A,
                                                const bf16* __restrict__ Bt,
                                                float* __restrict__ C) {
  const int lane = threadIdx.x & 31;
  const int wid = threadIdx.x >> 5;
  const int m0 = blockIdx.x * 16;
  const int n0 = (blockIdx.y * 4 + wid) * 64;
  v8f c0 = {}, c1 = {}, c2 = {}, c3 = {};
  for (int k0 = 0; k0 < NZ; k0 += 32) {
    v16bf a  = load_frag_k32(A, NZ, m0, k0, lane);
    v16bf b0 = load_frag_k32(Bt, NZ, n0,      k0, lane);
    v16bf b1 = load_frag_k32(Bt, NZ, n0 + 16, k0, lane);
    v16bf b2 = load_frag_k32(Bt, NZ, n0 + 32, k0, lane);
    v16bf b3 = load_frag_k32(Bt, NZ, n0 + 48, k0, lane);
    c0 = WMMA_BF16(a, b0, c0);
    c1 = WMMA_BF16(a, b1, c1);
    c2 = WMMA_BF16(a, b2, c2);
    c3 = WMMA_BF16(a, b3, c3);
  }
  const int half = (lane >> 4) & 1, lrow = lane & 15;
#pragma unroll
  for (int r = 0; r < 8; ++r) {
    size_t row = (size_t)(m0 + half * 8 + r);
    C[row * DM + n0 +      lrow] = c0[r];
    C[row * DM + n0 + 16 + lrow] = c1[r];
    C[row * DM + n0 + 32 + lrow] = c2[r];
    C[row * DM + n0 + 48 + lrow] = c3[r];
  }
}

// ---------------------------------------------------------------------------
extern "C" void kernel_launch(void* const* d_in, const int* in_sizes, int n_in,
                              void* d_out, int out_size, void* d_ws, size_t ws_size,
                              hipStream_t stream) {
  const float* x  = (const float*)d_in[0];
  const float* WQ = (const float*)d_in[1];
  const float* WK = (const float*)d_in[2];
  const float* WV = (const float*)d_in[3];
  const float* WO = (const float*)d_in[4];
  float* out = (float*)d_out;

  char* ws = (char*)d_ws;
  const size_t xbf_off  = 0;                                  // 8192*2048 bf16
  const size_t bqkv_off = xbf_off  + (size_t)ROWS * DM * 2;   // 1536*2048 bf16
  const size_t qkv_off  = bqkv_off + (size_t)NQKV * DM * 2;   // 8192*1536 bf16
  const size_t z_off    = qkv_off  + (size_t)ROWS * NQKV * 2; // 8192*512  bf16
  const size_t bo_off   = z_off    + (size_t)ROWS * NZ * 2;   // 2048*512  bf16
  bf16* xbf  = (bf16*)(ws + xbf_off);
  bf16* bqkv = (bf16*)(ws + bqkv_off);
  bf16* qkv  = (bf16*)(ws + qkv_off);
  bf16* z    = (bf16*)(ws + z_off);
  bf16* bo   = (bf16*)(ws + bo_off);

  const size_t nx = (size_t)ROWS * DM;
  cast_x_bf16<<<(unsigned)((nx + 255) / 256), 256, 0, stream>>>(x, xbf, nx);
  build_bqkv<<<(NQKV * DM + 255) / 256, 256, 0, stream>>>(WQ, WK, WV, bqkv);
  build_bo<<<(DM * NZ + 255) / 256, 256, 0, stream>>>(WO, bo);

  gemm_qkv<<<dim3(ROWS / 16, NQKV / 256), 128, 0, stream>>>(xbf, bqkv, qkv);
  flash_attn<<<dim3(S / 16 / 4, NKV, B), 128, 0, stream>>>(qkv, z);
  gemm_out<<<dim3(ROWS / 16, DM / 256), 128, 0, stream>>>(z, bo, out);
}
